// CotLayer_22325240005123
// MI455X (gfx1250) — matrix operand order repaired and checked by reading, not compiled, not run on hardware
//
#include <hip/hip_runtime.h>
#include <cmath>

// ---------------------------------------------------------------------------
// CotLayer on gfx1250: implicit-GEMM 3x3x3 convs on V_WMMA_F32_16X16X32_F16.
// Round 3: f16 activations stored with a zero halo ([b][t+1][h+1][w+1][C],
// padded 10x98x98) so every tap load is unconditional: no clamps, no
// cndmask zero-fill, tap offset is one wave-uniform SALU delta. Inner loop
// is now loads + back-to-back WMMA bursts only.
// ---------------------------------------------------------------------------

#define EPSV 1e-5f
constexpr int Bn = 2, Cc = 128, Tt = 8, Hh = 96, Ww = 96;
constexpr int HW = Hh * Ww;          // 9216
constexpr int PP = Tt * HW;          // 73728 positions per batch image
constexpr int NPOS = Bn * PP;        // 147456 total positions
constexpr int Tp = Tt + 2, Hp = Hh + 2, Wp = Ww + 2;   // padded dims
constexpr int PPAD = Tp * Hp * Wp;   // 96040 padded positions per image
constexpr int NPPAD = Bn * PPAD;     // 192080
constexpr int QKC = 256;             // channels of concat [x | kf]
constexpr int EMBC = 144;            // e2 output channels
constexpr int ATTNC = 64;

typedef __attribute__((ext_vector_type(16))) _Float16 v16h;
typedef __attribute__((ext_vector_type(8)))  _Float16 v8h;
typedef __attribute__((ext_vector_type(8)))  float    v8f;

__device__ __forceinline__ int ppos_of(int b, int t, int h, int w) {
  return ((b * Tp + t + 1) * Hp + (h + 1)) * Wp + (w + 1);
}

// ------------------------------ prep kernels -------------------------------

__global__ void prep_sb(const float* __restrict__ g, const float* __restrict__ b,
                        const float* __restrict__ m, const float* __restrict__ v,
                        float* __restrict__ s, float* __restrict__ t, int n) {
  int i = blockIdx.x * blockDim.x + threadIdx.x;
  if (i < n) {
    float sv = g[i] * rsqrtf(v[i] + EPSV);
    s[i] = sv;
    t[i] = b[i] - m[i] * sv;
  }
}

__global__ void fill_sb(const float* __restrict__ bias,
                        float* __restrict__ s, float* __restrict__ t, int n) {
  int i = blockIdx.x * blockDim.x + threadIdx.x;
  if (i < n) { s[i] = 1.0f; t[i] = bias[i]; }
}

__global__ void zero_f32(float* __restrict__ p, int n) {
  int i = blockIdx.x * blockDim.x + threadIdx.x;
  if (i < n) p[i] = 0.0f;
}

// zero n 16-byte chunks (halo clear for padded activation buffers)
__global__ void zero_b128(float4* __restrict__ p, long n) {
  long i = (long)blockIdx.x * blockDim.x + threadIdx.x;
  if (i < n) p[i] = make_float4(0.f, 0.f, 0.f, 0.f);
}

// OIDHW f32 -> [tap][O][I] f16
__global__ void pack_w(const float* __restrict__ w, _Float16* __restrict__ out,
                       int O, int I) {
  int total = O * I * 27;
  int idx = blockIdx.x * blockDim.x + threadIdx.x;
  if (idx >= total) return;
  int o = idx / (I * 27);
  int r = idx - o * I * 27;
  int i = r / 27;
  int tap = r - i * 27;
  out[((long)tap * O + o) * I + i] = (_Float16)w[idx];
}

// x NCDHW f32 -> padded channels-last f16 qk, channels [0,128)
__global__ void pack_x(const float* __restrict__ x, _Float16* __restrict__ qk) {
  long idx = (long)blockIdx.x * blockDim.x + threadIdx.x;   // over NPOS*128
  int c = (int)(idx & 127);
  long pos = idx >> 7;
  int b = (int)(pos / PP);
  int r = (int)(pos - (long)b * PP);
  int t = r / HW;
  int s = r - t * HW;
  int h = s / Ww, w = s - (s / Ww) * Ww;
  qk[(long)ppos_of(b, t, h, w) * QKC + c] = (_Float16)x[(long)(b * Cc + c) * PP + r];
}

// ------------------------- WMMA implicit-GEMM conv -------------------------
// One wave computes a 16(OC) x 64(pos) tile. Input is halo-padded: every tap
// load is valid, tap offset is a wave-uniform delta (SALU). Four B fragments
// are staged, then four back-to-back v_wmma (A reused, disjoint B/D).
template <int ICG, int ACTC>
__global__ void __launch_bounds__(128)
conv3d_wmma(const _Float16* __restrict__ act,
            const _Float16* __restrict__ wpack, int OC,
            int icPerGroup, int ocPerGroup, int chanOff,
            const float* __restrict__ scale, const float* __restrict__ bias,
            void* __restrict__ outp, int outC, int outOff,
            int outPadded, int outHalf, int relu) {
  const int lane = threadIdx.x & 31;
  const int wave = threadIdx.x >> 5;
  const int posBase = (blockIdx.x * (blockDim.x >> 5) + wave) * 64;
  const int ocBase = blockIdx.y * 16;
  const bool hi = lane >= 16;
  const int n = lane & 15;
  const int icb = (ocBase / ocPerGroup) * icPerGroup + chanOff;

  // per-lane padded center offsets for the 4 position sub-tiles
  int vofs[4], pposa[4];
#pragma unroll
  for (int j = 0; j < 4; ++j) {
    int p = posBase + j * 16 + n;
    int w = p % Ww; p /= Ww;
    int h = p % Hh; p /= Hh;
    int t = p % Tt; int b = p / Tt;
    pposa[j] = ppos_of(b, t, h, w);
    vofs[j] = pposa[j] * ACTC + icb + (hi ? 16 : 0);
  }

  v8f acc[4] = {};
  const _Float16* aBase = wpack + (long)(ocBase + n) * ICG + (hi ? 16 : 0);

  for (int dt = 0; dt < 3; ++dt)
    for (int dh = 0; dh < 3; ++dh)
      for (int dw = 0; dw < 3; ++dw) {
        const int tap = (dt * 3 + dh) * 3 + dw;
        // wave-uniform tap delta into the padded tensor (SALU only)
        const _Float16* tb =
            act + (long)((dt - 1) * (Hp * Wp) + (dh - 1) * Wp + (dw - 1)) * ACTC;
        const _Float16* ap = aBase + (long)tap * OC * ICG;
        if (tap < 26) __builtin_prefetch(ap + (long)OC * ICG, 0, 0);

#pragma unroll
        for (int kb = 0; kb < ICG; kb += 32) {
          // A fragment: lanes 0-15 K{kb..kb+15}, lanes 16-31 +16
          v8h alo = *(const v8h*)(ap + kb);
          v8h ahi = *(const v8h*)(ap + kb + 8);
          v16h a;
#pragma unroll
          for (int q = 0; q < 8; ++q) { a[q] = alo[q]; a[8 + q] = ahi[q]; }

          // stage all four B fragments (unconditional, halo-padded)
          v16h bfs[4];
#pragma unroll
          for (int j = 0; j < 4; ++j)
            bfs[j] = *(const v16h*)(tb + vofs[j] + kb);

          // back-to-back WMMA burst
#pragma unroll
          for (int j = 0; j < 4; ++j)
            acc[j] = __builtin_amdgcn_wmma_f32_16x16x32_f16(
                false, a, false, bfs[j], (short)0, acc[j], false, false);
        }
      }

  // epilogue: D row = i + (hi?8:0), col = lane&15; fused scale/bias/ReLU
#pragma unroll
  for (int j = 0; j < 4; ++j) {
#pragma unroll
    for (int i = 0; i < 8; ++i) {
      int oc = ocBase + i + (hi ? 8 : 0);
      float v = acc[j][i] * scale[oc] + bias[oc];
      if (relu) v = v > 0.0f ? v : 0.0f;
      long o = outPadded ? (long)pposa[j] * outC + outOff + oc
                         : (long)(posBase + j * 16 + n) * outC + outOff + oc;
      if (outHalf) ((_Float16*)outp)[o] = (_Float16)v;
      else         ((float*)outp)[o] = v;
    }
  }
}

// ------------------------------ GroupNorm ----------------------------------

__global__ void gn_reduce(const float* __restrict__ e2o, float* __restrict__ gstat) {
  int gid = blockIdx.x;        // 1024 blocks: 32 (b,g) * 32 chunks
  int bg = gid >> 5;
  int chunk = gid & 31;
  int b = bg >> 4, g = bg & 15;
  float s = 0.f, q = 0.f;
  for (int p = threadIdx.x; p < PP / 32; p += 256) {
    long pos = (long)b * PP + chunk * (PP / 32) + p;
    const float* row = e2o + pos * EMBC + g * 9;
#pragma unroll
    for (int cc = 0; cc < 9; ++cc) { float v = row[cc]; s += v; q += v * v; }
  }
  __shared__ float ss[256], sq[256];
  ss[threadIdx.x] = s; sq[threadIdx.x] = q;
  __syncthreads();
  for (int st = 128; st > 0; st >>= 1) {
    if (threadIdx.x < st) { ss[threadIdx.x] += ss[threadIdx.x + st];
                            sq[threadIdx.x] += sq[threadIdx.x + st]; }
    __syncthreads();
  }
  if (threadIdx.x == 0) {
    atomicAdd(&gstat[bg * 2 + 0], ss[0]);
    atomicAdd(&gstat[bg * 2 + 1], sq[0]);
  }
}

__global__ void gn_apply(float* __restrict__ e2o, const float* __restrict__ gstat,
                         const float* __restrict__ gg, const float* __restrict__ gb) {
  long idx = (long)blockIdx.x * blockDim.x + threadIdx.x;   // NPOS*144
  if (idx >= (long)NPOS * EMBC) return;
  int c = (int)(idx % EMBC);
  long pos = idx / EMBC;
  int b = (int)(pos / PP);
  int bg = b * 16 + c / 9;
  const float cnt = 9.0f * (float)PP;
  float mu = gstat[bg * 2 + 0] / cnt;
  float var = gstat[bg * 2 + 1] / cnt - mu * mu;
  e2o[idx] = (e2o[idx] - mu) * rsqrtf(var + EPSV) * gg[c] + gb[c];
}

// --------------- spatially-varying local conv + BN2 + swish ----------------

__global__ void local_conv(const float* __restrict__ xq, const float* __restrict__ wv,
                           const float* __restrict__ s2, const float* __restrict__ t2,
                           float* __restrict__ y) {
  long idx = (long)blockIdx.x * blockDim.x + threadIdx.x;   // NPOS*128
  int c = (int)(idx & 127);
  long pos = idx >> 7;
  int b = (int)(pos / PP);
  int r = (int)(pos - (long)b * PP);
  int t = r / HW;
  int s = r - t * HW;
  int h = s / Ww, w = s - (s / Ww) * Ww;
  const float* wvp = wv + pos * EMBC + (c >> 3) * 9;
  float acc = 0.f;
  int kk = 0;
  for (int dh = -1; dh <= 1; ++dh)
    for (int dw = -1; dw <= 1; ++dw, ++kk) {
      int hs = h + dh, ws = w + dw;
      if ((unsigned)hs < (unsigned)Hh && (unsigned)ws < (unsigned)Ww) {
        long np = (long)b * PP + t * HW + hs * Ww + ws;
        acc += xq[np * Cc + c] * wvp[kk];
      }
    }
  float yv = acc * s2[c] + t2[c];
  y[idx] = yv / (1.0f + expf(-yv));   // swish
}

// --------------------------- gap over (H,W) of y+kf ------------------------

__global__ void gap_reduce(const float* __restrict__ y, const _Float16* __restrict__ qk,
                           float* __restrict__ gap) {
  int gid = blockIdx.x;          // 2048: [b][t][c]
  int c = gid & 127;
  int t = (gid >> 7) & 7;
  int b = gid >> 10;
  float sum = 0.f;
  for (int s = threadIdx.x; s < HW; s += 256) {
    int h = s / Ww, w = s - (s / Ww) * Ww;
    long pos = (long)b * PP + t * HW + s;
    sum += y[pos * Cc + c] +
           (float)qk[(long)ppos_of(b, t, h, w) * QKC + Cc + c];
  }
  __shared__ float ss[256];
  ss[threadIdx.x] = sum;
  __syncthreads();
  for (int st = 128; st > 0; st >>= 1) {
    if (threadIdx.x < st) ss[threadIdx.x] += ss[threadIdx.x + st];
    __syncthreads();
  }
  if (threadIdx.x == 0) gap[(b * Cc + c) * Tt + t] = ss[0] / (float)HW;
}

// --------------------- SE path (tiny) + radix softmax ----------------------
// conv3d on [B,C,T,1,1] with pad 1: only the center (dh=dw=1) spatial tap of
// the 3x3x3 kernel contributes; it reduces to a kernel-3 conv over T.
__global__ void se_attn(const float* __restrict__ gap,
                        const float* __restrict__ w1, const float* __restrict__ b1,
                        const float* __restrict__ g1, const float* __restrict__ bb1,
                        const float* __restrict__ m1, const float* __restrict__ v1,
                        const float* __restrict__ w2, const float* __restrict__ b2,
                        float* __restrict__ attnP) {
  __shared__ float a1s[Bn * ATTNC * Tt];       // 1024
  __shared__ float atts[Bn * 2 * Cc * Tt];     // 4096
  for (int i = threadIdx.x; i < Bn * ATTNC * Tt; i += blockDim.x) {
    int b = i >> 9, o = (i >> 3) & 63, t = i & 7;
    float sum = b1[o];
    for (int dt = 0; dt < 3; ++dt) {
      int tt = t + dt - 1;
      if ((unsigned)tt >= (unsigned)Tt) continue;
      const float* wr = w1 + ((long)o * Cc) * 27 + dt * 9 + 4;
      for (int c = 0; c < Cc; ++c)
        sum += wr[(long)c * 27] * gap[(b * Cc + c) * Tt + tt];
    }
    float sc = g1[o] * rsqrtf(v1[o] + EPSV);
    float a = (sum - m1[o]) * sc + bb1[o];
    a1s[i] = a > 0.f ? a : 0.f;
  }
  __syncthreads();
  for (int i = threadIdx.x; i < Bn * 2 * Cc * Tt; i += blockDim.x) {
    int b = i >> 11, o2 = (i >> 3) & 255, t = i & 7;
    float sum = b2[o2];
    for (int dt = 0; dt < 3; ++dt) {
      int tt = t + dt - 1;
      if ((unsigned)tt >= (unsigned)Tt) continue;
      const float* wr = w2 + ((long)o2 * ATTNC) * 27 + dt * 9 + 4;
      for (int o = 0; o < ATTNC; ++o)
        sum += wr[(long)o * 27] * a1s[(b << 9) | (o << 3) | tt];
    }
    atts[i] = sum;
  }
  __syncthreads();
  for (int i = threadIdx.x; i < Bn * Cc * Tt; i += blockDim.x) {
    int b = i >> 10, c = (i >> 3) & 127, t = i & 7;
    float l0 = atts[(b << 11) | ((2 * c) << 3) | t];
    float l1 = atts[(b << 11) | ((2 * c + 1) << 3) | t];
    float mx = fmaxf(l0, l1);
    float e0 = expf(l0 - mx), e1 = expf(l1 - mx);
    float inv = 1.0f / (e0 + e1);
    attnP[((b * Cc + c) * 2 + 0) * Tt + t] = e0 * inv;
    attnP[((b * Cc + c) * 2 + 1) * Tt + t] = e1 * inv;
  }
}

// ----------------------------- final blend ---------------------------------

__global__ void final_blend(const float* __restrict__ y, const _Float16* __restrict__ qk,
                            const float* __restrict__ attnP, float* __restrict__ out) {
  long idx = (long)blockIdx.x * blockDim.x + threadIdx.x;   // NCDHW order
  int s = (int)(idx % HW);
  long r = idx / HW;
  int t = (int)(r % Tt); r /= Tt;
  int c = (int)(r % Cc);
  int b = (int)(r / Cc);
  int h = s / Ww, w = s - (s / Ww) * Ww;
  long pos = (long)b * PP + t * HW + s;
  float yv = y[pos * Cc + c];
  float kv = (float)qk[(long)ppos_of(b, t, h, w) * QKC + Cc + c];
  float p0 = attnP[((b * Cc + c) * 2 + 0) * Tt + t];
  float p1 = attnP[((b * Cc + c) * 2 + 1) * Tt + t];
  out[idx] = yv * p0 + kv * p1;
}

// ------------------------------- launcher ----------------------------------

extern "C" void kernel_launch(void* const* d_in, const int* in_sizes, int n_in,
                              void* d_out, int out_size, void* d_ws, size_t ws_size,
                              hipStream_t stream) {
  const float* x       = (const float*)d_in[0];
  const float* ke_w    = (const float*)d_in[1];
  const float* ke_g    = (const float*)d_in[2];
  const float* ke_b    = (const float*)d_in[3];
  const float* ke_m    = (const float*)d_in[4];
  const float* ke_v    = (const float*)d_in[5];
  const float* e1_w    = (const float*)d_in[6];
  const float* e1_g    = (const float*)d_in[7];
  const float* e1_b    = (const float*)d_in[8];
  const float* e1_m    = (const float*)d_in[9];
  const float* e1_v    = (const float*)d_in[10];
  const float* e2_w    = (const float*)d_in[11];
  const float* e2_bias = (const float*)d_in[12];
  const float* gn_g    = (const float*)d_in[13];
  const float* gn_b    = (const float*)d_in[14];
  const float* c1_w    = (const float*)d_in[15];
  const float* c1_g    = (const float*)d_in[16];
  const float* c1_b    = (const float*)d_in[17];
  const float* c1_m    = (const float*)d_in[18];
  const float* c1_v    = (const float*)d_in[19];
  const float* bn2_g   = (const float*)d_in[20];
  const float* bn2_b   = (const float*)d_in[21];
  const float* bn2_m   = (const float*)d_in[22];
  const float* bn2_v   = (const float*)d_in[23];
  const float* se1_w   = (const float*)d_in[24];
  const float* se1_bias= (const float*)d_in[25];
  const float* se1_g   = (const float*)d_in[26];
  const float* se1_b   = (const float*)d_in[27];
  const float* se1_m   = (const float*)d_in[28];
  const float* se1_v   = (const float*)d_in[29];
  const float* se2_w   = (const float*)d_in[30];
  const float* se2_bias= (const float*)d_in[31];
  (void)in_sizes; (void)n_in; (void)out_size; (void)ws_size;

  char* ws = (char*)d_ws;
  size_t off = 0;
  auto take = [&](size_t bytes) -> char* {
    char* p = ws + off;
    off += (bytes + 255) & ~(size_t)255;
    return p;
  };
  const size_t qkBytes  = (size_t)NPPAD * QKC * 2;   // padded [ppos][256]
  const size_t wv1Bytes = (size_t)NPPAD * 64 * 2;    // padded [ppos][64]
  _Float16* qk   = (_Float16*)take(qkBytes);
  _Float16* wv1  = (_Float16*)take(wv1Bytes);
  float*    e2o  = (float*)   take((size_t)NPOS * EMBC * 4);  // unpadded
  float*    xq   = (float*)   take((size_t)NPOS * Cc * 4);    // unpadded
  float*    yb   = (float*)   take((size_t)NPOS * Cc * 4);    // unpadded
  _Float16* Wke  = (_Float16*)take((size_t)27 * 128 * 32 * 2);
  _Float16* We1  = (_Float16*)take((size_t)27 * 64 * 256 * 2);
  _Float16* We2  = (_Float16*)take((size_t)27 * 144 * 64 * 2);
  _Float16* Wc1  = (_Float16*)take((size_t)27 * 128 * 128 * 2);
  float* s_ke  = (float*)take(128 * 4);  float* t_ke  = (float*)take(128 * 4);
  float* s_e1  = (float*)take(64 * 4);   float* t_e1  = (float*)take(64 * 4);
  float* s_e2  = (float*)take(144 * 4);  float* t_e2  = (float*)take(144 * 4);
  float* s_c1  = (float*)take(128 * 4);  float* t_c1  = (float*)take(128 * 4);
  float* s_bn2 = (float*)take(128 * 4);  float* t_bn2 = (float*)take(128 * 4);
  float* gstat = (float*)take(64 * 4);
  float* gapB  = (float*)take(2048 * 4);
  float* attnP = (float*)take(4096 * 4);

  // halo clear (whole padded buffers; interiors overwritten below)
  {
    long n16 = (long)(qkBytes / 16);
    zero_b128<<<(int)((n16 + 255) / 256), 256, 0, stream>>>((float4*)qk, n16);
    n16 = (long)(wv1Bytes / 16);
    zero_b128<<<(int)((n16 + 255) / 256), 256, 0, stream>>>((float4*)wv1, n16);
  }

  // BN folds + weight/activation packing
  prep_sb<<<1, 128, 0, stream>>>(ke_g, ke_b, ke_m, ke_v, s_ke, t_ke, 128);
  prep_sb<<<1, 128, 0, stream>>>(e1_g, e1_b, e1_m, e1_v, s_e1, t_e1, 64);
  prep_sb<<<1, 128, 0, stream>>>(c1_g, c1_b, c1_m, c1_v, s_c1, t_c1, 128);
  prep_sb<<<1, 128, 0, stream>>>(bn2_g, bn2_b, bn2_m, bn2_v, s_bn2, t_bn2, 128);
  fill_sb<<<1, 256, 0, stream>>>(e2_bias, s_e2, t_e2, 144);

  pack_w<<<(128 * 32 * 27 + 255) / 256, 256, 0, stream>>>(ke_w, Wke, 128, 32);
  pack_w<<<(64 * 256 * 27 + 255) / 256, 256, 0, stream>>>(e1_w, We1, 64, 256);
  pack_w<<<(144 * 64 * 27 + 255) / 256, 256, 0, stream>>>(e2_w, We2, 144, 64);
  pack_w<<<(128 * 128 * 27 + 255) / 256, 256, 0, stream>>>(c1_w, Wc1, 128, 128);
  pack_x<<<NPOS * Cc / 256, 256, 0, stream>>>(x, qk);

  const int gx = NPOS / 256;   // 576 blocks of 4 waves * 64 positions
  // ke: grouped 128->128 (4 groups of 32), ReLU, write kf into qk[...,128:)
  conv3d_wmma<32, QKC><<<dim3(gx, 8), 128, 0, stream>>>(
      qk, Wke, 128, 32, 32, 0, s_ke, t_ke, qk, QKC, 128, 1, 1, 1);
  // e1: 256->64, ReLU -> wv1 (padded f16)
  conv3d_wmma<256, QKC><<<dim3(gx, 4), 128, 0, stream>>>(
      qk, We1, 64, 0, 64, 0, s_e1, t_e1, wv1, 64, 0, 1, 1, 1);
  // e2: 64->144 (+bias) -> e2o (f32, unpadded, pre-GN)
  conv3d_wmma<64, 64><<<dim3(gx, 9), 128, 0, stream>>>(
      wv1, We2, 144, 0, 144, 0, s_e2, t_e2, e2o, EMBC, 0, 0, 0, 0);
  // c1: 128->128 (BN folded, no ReLU) reads x half of qk -> xq (f32 unpadded)
  conv3d_wmma<128, QKC><<<dim3(gx, 8), 128, 0, stream>>>(
      qk, Wc1, 128, 0, 128, 0, s_c1, t_c1, xq, Cc, 0, 0, 0, 0);

  // GroupNorm (16 groups of 9 channels) in-place on e2o
  zero_f32<<<1, 64, 0, stream>>>(gstat, 64);
  gn_reduce<<<1024, 256, 0, stream>>>(e2o, gstat);
  gn_apply<<<((long)NPOS * EMBC + 255) / 256, 256, 0, stream>>>(e2o, gstat, gn_g, gn_b);

  // spatially-varying local conv + BN2 + swish -> y
  local_conv<<<NPOS * Cc / 256, 256, 0, stream>>>(xq, e2o, s_bn2, t_bn2, yb);

  // gap over (H,W) of (y + kf)
  gap_reduce<<<Bn * Cc * Tt, 256, 0, stream>>>(yb, qk, gapB);

  // SE convs + radix-2 softmax (tiny)
  se_attn<<<1, 256, 0, stream>>>(gapB, se1_w, se1_bias, se1_g, se1_b, se1_m, se1_v,
                                 se2_w, se2_bias, attnP);

  // blend to NCDHW f32 output
  final_blend<<<NPOS * Cc / 256, 256, 0, stream>>>(yb, qk, attnP, (float*)d_out);
}
